// Seq2seq_66391604461892
// MI455X (gfx1250) — compile-verified, matrix operand order checked by reading, not run on hardware
//
#include <hip/hip_runtime.h>
#include <stdint.h>

typedef __attribute__((ext_vector_type(16))) __bf16 v16bf;
typedef __attribute__((ext_vector_type(8)))  float  v8f;
typedef unsigned short u16;

#define DEV static __device__ __forceinline__

DEV u16 f2bf(float f){
  union { float f; unsigned u; } x; x.f = f;
  unsigned r = x.u + 0x7FFFu + ((x.u >> 16) & 1u);
  return (u16)(r >> 16);
}
DEV float sigf(float x){ return 1.0f/(1.0f+__expf(-x)); }
DEV int uni(int x){ return __builtin_amdgcn_readfirstlane(x); }

// Load a 16x32 bf16 fragment (A or column-major B) per CDNA5 ISA layout:
// lane<16: rows 0..15, K chunks [0,8)+[16,24); lane>=16: chunks [8,16)+[24,32).
DEV v16bf load_frag(const u16* base, int ld, int lane){
  int r   = lane & 15;
  int off = (lane >> 4) << 3;
  const u16* p = base + (long)r*ld + off;
  union { uint4 q[2]; v16bf v; } u;
  u.q[0] = *(const uint4*)(p);
  u.q[1] = *(const uint4*)(p + 16);
  return u.v;
}
DEV v8f wmma_bf(v16bf a, v16bf b, v8f c){
  return __builtin_amdgcn_wmma_f32_16x16x32_bf16(false, a, false, b, (short)0, c, false, false);
}
DEV void store_tile(float* C, v8f acc, int r0, int nt, int cb, int N,
                    const float* bias, int act){
  int col = nt*16 + cb;
  if (col < N){
    float bv = bias ? bias[col] : 0.0f;
    #pragma unroll
    for (int v=0; v<8; v++){
      float val = acc[v] + bv;
      if (act == 1) val = tanhf(val);
      C[(long)(r0+v)*N + col] = val;
    }
  }
}

// ---------------- generic batched bf16 GEMM:  C[M,N] = act(A[M,Kp] * W[N,Kp]^T + bias)
// Wave tile: 32 rows x 64 cols (2 A-frags x 4 B-frags = 8 WMMAs per k-step).
// GUARD=false => N multiple of 64, branch-free inner loop (EXEC untouched around WMMA).
template<bool GUARD>
__global__ void __launch_bounds__(256) gemm_bf16_nt(
    const u16* __restrict__ A, long sA,
    const u16* __restrict__ W, long sW,
    const float* __restrict__ bias,
    float* __restrict__ C, long sC,
    int Mr, int N, int Kp, int act)
{
  int lane = threadIdx.x & 31, wave = threadIdx.x >> 5;
  const u16* Ab = A + (long)blockIdx.z * sA;
  const u16* Wb = W + (long)blockIdx.z * sW;
  float*     Cb = C + (long)blockIdx.z * sC;
  int row0 = blockIdx.y << 5;
  int nt0  = (blockIdx.x * 8 + wave) << 2;
  if (uni(nt0*16 >= N)) return;               // scalar branch, whole-wave
  int e1 = 1, e2 = 1, e3 = 1;
  if (GUARD){
    e1 = uni((nt0+1)*16 < N);
    e2 = uni((nt0+2)*16 < N);
    e3 = uni((nt0+3)*16 < N);
  }
  v8f z8 = {0,0,0,0,0,0,0,0};
  v8f c00=z8,c01=z8,c10=z8,c11=z8,c20=z8,c21=z8,c30=z8,c31=z8;
  const u16* Ar0 = Ab + (long)row0 * Kp;
  const u16* Ar1 = Ar0 + (long)16 * Kp;
  const u16* W0 = Wb + (long)(nt0+0)*16*Kp;
  const u16* W1 = Wb + (long)(nt0+1)*16*Kp;
  const u16* W2 = Wb + (long)(nt0+2)*16*Kp;
  const u16* W3 = Wb + (long)(nt0+3)*16*Kp;
  // software pipeline: A-frags for k prefetched one iteration ahead
  v16bf a0 = load_frag(Ar0, Kp, lane);
  v16bf a1 = load_frag(Ar1, Kp, lane);
  for (int k=0; k<Kp; k+=32){
    __builtin_prefetch(W0 + k + 256, 0, 1);
    // issue all B loads for this k before the first WMMA
    v16bf b0 = load_frag(W0 + k, Kp, lane);
    v16bf b1 = b0, b2 = b0, b3 = b0;
    if (!GUARD || e1) b1 = load_frag(W1 + k, Kp, lane);
    if (!GUARD || e2) b2 = load_frag(W2 + k, Kp, lane);
    if (!GUARD || e3) b3 = load_frag(W3 + k, Kp, lane);
    v16bf a0n = a0, a1n = a1;
    if (k + 32 < Kp){
      a0n = load_frag(Ar0 + k + 32, Kp, lane);
      a1n = load_frag(Ar1 + k + 32, Kp, lane);
    }
    c00 = wmma_bf(a0, b0, c00); c01 = wmma_bf(a1, b0, c01);
    if (!GUARD || e1){ c10 = wmma_bf(a0, b1, c10); c11 = wmma_bf(a1, b1, c11); }
    if (!GUARD || e2){ c20 = wmma_bf(a0, b2, c20); c21 = wmma_bf(a1, b2, c21); }
    if (!GUARD || e3){ c30 = wmma_bf(a0, b3, c30); c31 = wmma_bf(a1, b3, c31); }
    a0 = a0n; a1 = a1n;
  }
  int r0 = row0 + ((lane>>4)<<3);
  int cb = lane & 15;
  store_tile(Cb, c00, r0,    nt0+0, cb, N, bias, act);
  store_tile(Cb, c01, r0+16, nt0+0, cb, N, bias, act);
  if (!GUARD || e1){ store_tile(Cb, c10, r0, nt0+1, cb, N, bias, act);
                     store_tile(Cb, c11, r0+16, nt0+1, cb, N, bias, act); }
  if (!GUARD || e2){ store_tile(Cb, c20, r0, nt0+2, cb, N, bias, act);
                     store_tile(Cb, c21, r0+16, nt0+2, cb, N, bias, act); }
  if (!GUARD || e3){ store_tile(Cb, c30, r0, nt0+3, cb, N, bias, act);
                     store_tile(Cb, c31, r0+16, nt0+3, cb, N, bias, act); }
}

// ---------------- pack fp32 -> bf16 with column offset + zero K padding
__global__ void pack_bf16_k(const float* __restrict__ src, int lds, int scol, int K,
                            u16* __restrict__ dst, int ldd, int dcol, int Kp, long total)
{
  long i = (long)blockIdx.x*256 + threadIdx.x;
  if (i >= total) return;
  long r = i / Kp; int k = (int)(i % Kp);
  dst[r*ldd + dcol + k] = (k < K) ? f2bf(src[r*(long)lds + scol + k]) : (u16)0;
}

__global__ void add_vec_k(const float* a, const float* b, float* d, int n){
  int i = blockIdx.x*256 + threadIdx.x; if (i < n) d[i] = a[i] + b[i];
}
__global__ void fill_k(float* p, long n, float v){
  long i = (long)blockIdx.x*256 + threadIdx.x; if (i < n) p[i] = v;
}
__global__ void copy_cols_k(const float* s, int K, float* d, int ldd, int coloff, long total){
  long i = (long)blockIdx.x*256 + threadIdx.x; if (i >= total) return;
  long r = i / K; int k = (int)(i % K);
  d[r*ldd + coloff + k] = s[i];
}

// ---------------- graph static attention -> bf16 encoder embedding rows [8192,512]
__global__ void __launch_bounds__(128) graph_emb_k(
    const float* __restrict__ emb_enc, const float* __restrict__ tag_emb,
    const float* __restrict__ ent_emb, const float* __restrict__ rel_emb,
    const float* __restrict__ hW, const float* __restrict__ tW, const float* __restrict__ rW,
    const int* __restrict__ src, const int* __restrict__ tag,
    const int* __restrict__ cs, const int* __restrict__ mask,
    u16* __restrict__ emb_bf)
{
  __shared__ float hd[800], tl[800], rl[800];
  __shared__ float beta[8], alpha[8];
  int tok = blockIdx.x;
  const int* csr = cs + (long)tok*24;
  const int* mr  = mask + (long)tok*8;
  for (int i=threadIdx.x; i<800; i+=128){
    int m = i/100, k = i%100;
    int msk = mr[m];
    hd[i] = msk ? 0.f : ent_emb[(long)csr[m*3+0]*100 + k];
    rl[i] = msk ? 0.f : rel_emb[(long)csr[m*3+1]*100 + k];
    tl[i] = msk ? 0.f : ent_emb[(long)csr[m*3+2]*100 + k];
  }
  if (threadIdx.x < 8) beta[threadIdx.x] = 0.f;
  __syncthreads();
  for (int i=threadIdx.x; i<800; i+=128){
    int m = i/100, d = i%100;
    float hw=0.f, tw=0.f, rw=0.f;
    for (int k=0;k<100;k++){
      hw += hd[m*100+k]*hW[d*100+k];
      tw += tl[m*100+k]*tW[d*100+k];
      rw += rl[m*100+k]*rW[d*100+k];
    }
    atomicAdd(&beta[m], rw * tanhf(hw + tw));
  }
  __syncthreads();
  if (threadIdx.x == 0){
    float mx = -3.0e38f;
    for (int m=0;m<8;m++){ if (mr[m]) beta[m] = -1e12f; mx = fmaxf(mx, beta[m]); }
    float s = 0.f;
    for (int m=0;m<8;m++){ alpha[m] = __expf(beta[m]-mx); s += alpha[m]; }
    float inv = 1.0f/s;
    for (int m=0;m<8;m++) alpha[m] *= inv;
  }
  __syncthreads();
  u16* dst = emb_bf + (long)tok*512;
  int sv = src[tok], tg = tag[tok];
  for (int k=threadIdx.x; k<512; k+=128){
    float v;
    if (k < 300)      v = emb_enc[(long)sv*300 + k];
    else if (k < 303) v = tag_emb[tg*3 + (k-300)];
    else if (k < 503){
      int d = k-303; float s = 0.f;
      for (int m=0;m<8;m++) s += alpha[m]*(d<100 ? hd[m*100+d] : tl[m*100+d-100]);
      v = s;
    } else v = 0.f;
    dst[k] = f2bf(v);
  }
}

// ---------------- persistent 256-step LSTM scan (hid=512), h in LDS(bf16), c in VGPRs
// Each wave owns 16 N-tiles; B-frag shared across both M-tiles (2 WMMAs per B load).
__global__ void __launch_bounds__(256) lstm_scan_k(
    const float* __restrict__ xz,   // [B*T,2048] precomputed x@Wih^T+bias, rows b*T+t
    const u16*   __restrict__ Whh,  // [2048,512] bf16
    float* __restrict__ y, int yoff,// [B*T,1024], write at column yoff
    float* __restrict__ zbuf,       // [32,2048] scratch
    float* __restrict__ hfin, float* __restrict__ cfin, // [32,512]
    int T, int rev)
{
  __shared__ __align__(16) u16 hbf[32*512];
  int tid = threadIdx.x, lane = tid & 31, wave = tid >> 5;
  for (int i=tid; i<32*512; i+=256) hbf[i] = 0;
  float c[64];
  #pragma unroll
  for (int q=0;q<64;q++) c[q] = 0.f;
  int gb = tid >> 3, j0 = tid & 7;
  __syncthreads();
  const u16* a0base = (const u16*)hbf;
  const u16* a1base = (const u16*)hbf + 16*512;
  for (int s=0; s<T; s++){
    int t = rev ? (T-1-s) : s;
    for (int ni=0; ni<16; ni++){
      int nt = wave*16 + ni;
      const u16* wrow = Whh + (long)nt*16*512;
      v8f acc0 = {0,0,0,0,0,0,0,0}, acc1 = acc0;
      for (int k=0;k<512;k+=32){
        v16bf b  = load_frag(wrow + k, 512, lane);
        v16bf a0 = load_frag(a0base + k, 512, lane);
        v16bf a1 = load_frag(a1base + k, 512, lane);
        acc0 = wmma_bf(a0, b, acc0);
        acc1 = wmma_bf(a1, b, acc1);
      }
      int r0  = (lane>>4)<<3;
      int col = nt*16 + (lane&15);
      #pragma unroll
      for (int v=0;v<8;v++){
        zbuf[(r0+v)*2048 + col]      = acc0[v];
        zbuf[(16+r0+v)*2048 + col]   = acc1[v];
      }
    }
    __threadfence();
    __syncthreads();
    const float* xr = xz + ((long)gb*T + t)*2048;
    const float* zr = zbuf + gb*2048;
    float* yr = y + ((long)gb*T + t)*1024 + yoff;
    for (int q=0;q<64;q++){
      int j = j0 + (q<<3);
      float zi = zr[j]      + xr[j];
      float zf = zr[512+j]  + xr[512+j];
      float zg = zr[1024+j] + xr[1024+j];
      float zo = zr[1536+j] + xr[1536+j];
      float cn = sigf(zf)*c[q] + sigf(zi)*tanhf(zg);
      float hn = sigf(zo)*tanhf(cn);
      c[q] = cn;
      hbf[gb*512 + j] = f2bf(hn);
      yr[j] = hn;
      if (s == T-1){ hfin[gb*512+j] = hn; cfin[gb*512+j] = cn; }
    }
    __syncthreads();
  }
}

__global__ void __launch_bounds__(256) softmax_k(const float* __restrict__ src,
                                                 float* __restrict__ dst, int C){
  __shared__ float red[256];
  long row = blockIdx.x;
  const float* s = src + row*(long)C;
  float* d = dst + row*(long)C;
  int tid = threadIdx.x;
  float m = -3.0e38f;
  for (int j=tid;j<C;j+=256) m = fmaxf(m, s[j]);
  red[tid]=m; __syncthreads();
  for (int o=128;o;o>>=1){ if (tid<o) red[tid]=fmaxf(red[tid],red[tid+o]); __syncthreads(); }
  m = red[0]; __syncthreads();
  float sum = 0.f;
  for (int j=tid;j<C;j+=256){ float e=__expf(s[j]-m); d[j]=e; sum+=e; }
  red[tid]=sum; __syncthreads();
  for (int o=128;o;o>>=1){ if (tid<o) red[tid]+=red[tid+o]; __syncthreads(); }
  float inv = 1.0f/red[0];
  for (int j=tid;j<C;j+=256) d[j] *= inv;
}

__global__ void transpose_pack_k(const float* __restrict__ outp, u16* __restrict__ outT){
  long i = (long)blockIdx.x*256 + threadIdx.x;
  if (i >= (long)32*1024*256) return;
  int b = (int)(i >> 18);
  int r = (int)(i & 262143);
  int d = r >> 8, t = r & 255;
  outT[i] = f2bf(outp[((long)b*256 + t)*1024 + d]);
}

__global__ void combine_k(const float* ft, const float* gt, const float* y1, float* eo){
  long i = (long)blockIdx.x*256 + threadIdx.x;
  if (i >= (long)8192*1024) return;
  float f = tanhf(ft[i]); float g = sigf(gt[i]);
  eo[i] = g*f + (1.0f - g)*y1[i];
}

__global__ void lstm_gate_k(const float* __restrict__ z, float* __restrict__ c,
                            float* __restrict__ h, int hid, int n){
  int i = blockIdx.x*256 + threadIdx.x; if (i >= n) return;
  int b = i / hid, j = i % hid;
  const float* zr = z + (long)b*4*hid;
  float zi=zr[j], zf=zr[hid+j], zg=zr[2*hid+j], zo=zr[3*hid+j];
  float cn = sigf(zf)*c[i] + sigf(zi)*tanhf(zg);
  c[i] = cn; h[i] = sigf(zo)*tanhf(cn);
}

__global__ void pack_dec_in_k(const float* __restrict__ emb_dec, const int* __restrict__ trg,
                              int step, int Tt, const float* __restrict__ pctx,
                              u16* __restrict__ dst){
  int i = blockIdx.x*256 + threadIdx.x;
  if (i >= 32*1344) return;
  int b = i / 1344, k = i % 1344;
  float v;
  if (k < 300)      v = emb_dec[(long)trg[b*Tt + step]*300 + k];
  else if (k < 320) v = 0.f;
  else              v = pctx[b*1024 + (k-320)];
  dst[i] = f2bf(v);
}

__global__ void energy_k(const float* __restrict__ h1, const float* __restrict__ mems,
                         float* __restrict__ e){
  int gw = (blockIdx.x*256 + threadIdx.x) >> 5;
  int lane = threadIdx.x & 31;
  int b = gw >> 8, t = gw & 255;
  const float* m = mems + ((long)b*256 + t)*1024;
  const float* h = h1 + b*1024;
  float s = 0.f;
  for (int k=lane;k<1024;k+=32) s += h[k]*m[k];
  for (int o=16;o;o>>=1) s += __shfl_down(s, o, 32);
  if (!lane) e[b*256 + t] = s;
}

__global__ void ctx2_k(const float* __restrict__ attn, const float* __restrict__ mems,
                       float* __restrict__ ctx){
  int i = blockIdx.x*256 + threadIdx.x;
  if (i >= 32*1024) return;
  int b = i >> 10, d = i & 1023;
  const float* a = attn + b*256;
  const float* m = mems + (long)b*256*1024 + d;
  float s = 0.f;
  #pragma unroll 4
  for (int t=0;t<256;t++) s += a[t]*m[(long)t*1024];
  ctx[i] = s;
}

__global__ void extend_k(const float* __restrict__ lg, float* __restrict__ out, long total){
  long i = (long)blockIdx.x*256 + threadIdx.x; if (i >= total) return;
  long row = i / 50050; int j = (int)(i % 50050);
  int b = (int)(row / 31), st = (int)(row % 31);
  out[i] = (j < 50000) ? lg[((long)st*32 + b)*50000 + j] : 0.0f;
}

__global__ void __launch_bounds__(256) scat_k(const float* __restrict__ eAll,
                                              const int* __restrict__ ext,
                                              float* __restrict__ out){
  __shared__ float e[256]; __shared__ int pos[256];
  int row = blockIdx.x; int b = row/31, st = row%31;
  int t = threadIdx.x;
  e[t]   = eAll[((long)st*32 + b)*256 + t];
  pos[t] = ext[b*256 + t];
  __syncthreads();
  float mx = e[t]; bool win = true;
  for (int u=0;u<256;u++){
    if (pos[u] == pos[t]){
      float eu = e[u];
      if (eu > mx) mx = eu;
      if (eu > e[t] || (eu == e[t] && u < t)) win = false;
    }
  }
  if (win) out[(long)row*50050 + pos[t]] += mx;
}

__global__ void final_k(float* out, long total){
  long i = (long)blockIdx.x*256 + threadIdx.x; if (i >= total) return;
  float v = out[i];
  out[i] = (v == 0.0f) ? -1e12f : v;
}

// =======================================================================================
extern "C" void kernel_launch(void* const* d_in, const int* in_sizes, int n_in,
                              void* d_out, int out_size, void* d_ws, size_t ws_size,
                              hipStream_t stream)
{
  (void)in_sizes; (void)n_in; (void)out_size; (void)ws_size;
  const float* F[44]; for (int i=0;i<44;i++) F[i] = (const float*)d_in[i];
  const int* src = (const int*)d_in[44];
  const int* tag = (const int*)d_in[45];
  const int* cs  = (const int*)d_in[46];
  const int* msk = (const int*)d_in[47];
  const int* ext = (const int*)d_in[49];
  const int* trg = (const int*)d_in[50];
  float* out = (float*)d_out;

  char* wsp = (char*)d_ws;
  auto A8 = [&](long bytes)->char*{ char* r = wsp; wsp += (bytes + 255) & ~255L; return r; };
  auto AU = [&](long n)->u16*  { return (u16*)A8(n*2); };
  auto AF = [&](long n)->float*{ return (float*)A8(n*4); };

  // bf16 buffers
  u16* emb_bf  = AU((long)8192*512);
  u16* Wih0f   = AU(2048L*512); u16* Whh0f = AU(2048L*512);
  u16* Wih0b   = AU(2048L*512); u16* Whh0b = AU(2048L*512);
  u16* Wih1f   = AU(2048L*1024); u16* Whh1f = AU(2048L*512);
  u16* Wih1b   = AU(2048L*1024); u16* Whh1b = AU(2048L*512);
  u16* ltWp    = AU(1024L*1024); u16* updWp = AU(1024L*2048); u16* gateWp = AU(1024L*2048);
  u16* enctrWp = AU(1024L*1024); u16* redWp = AU(320L*1344);
  u16* Wc0p    = AU(4096L*1344); u16* Wc1p  = AU(4096L*2048);
  u16* catWp   = AU(1024L*2048); u16* logWp = AU(50000L*1024);
  u16* y0_bf   = AU((long)8192*1024); u16* out_bf = AU((long)8192*1024);
  u16* mem_bf  = AU((long)8192*1024); u16* sc_bf  = AU((long)8192*256);
  u16* outT_bf = AU((long)32*1024*256);
  u16* gin_bf  = AU((long)8192*2048); u16* eo_bf = AU((long)8192*1024);
  u16* Li_bf   = AU((long)992*1024);
  u16* xred_bf = AU(32L*1344); u16* xh0_bf = AU(32L*1344);
  u16* xh1_bf  = AU(32L*2048); u16* hc_bf  = AU(32L*2048);
  // fp32 buffers
  float* xz_a = AF((long)8192*2048); float* xz_b = AF((long)8192*2048);
  float* y0   = AF((long)8192*1024); float* y1   = AF((long)8192*1024);
  float* mem  = AF((long)8192*1024); float* en   = AF((long)8192*256);
  float* ctx  = AF((long)8192*1024); float* ft   = AF((long)8192*1024);
  float* gt   = AF((long)8192*1024); float* eo   = AF((long)8192*1024);
  float* mems = AF((long)8192*1024);
  float* zbuf = AF(32L*2048);
  float* h0ff = AF(32L*512); float* c0ff = AF(32L*512);
  float* h0bf = AF(32L*512); float* c0bf = AF(32L*512);
  float* h1ff = AF(32L*512); float* c1ff = AF(32L*512);
  float* h1bf = AF(32L*512); float* c1bf = AF(32L*512);
  float* bs0f = AF(2048); float* bs0b = AF(2048); float* bs1f = AF(2048); float* bs1b = AF(2048);
  float* bsd0 = AF(4096); float* bsd1 = AF(4096);
  float* dh0 = AF(32L*1024); float* dh1 = AF(32L*1024);
  float* dc0 = AF(32L*1024); float* dc1 = AF(32L*1024); float* pctx = AF(32L*1024);
  float* zdec = AF(32L*4096); float* xr = AF(32L*300);
  float* eAll = AF(31L*32*256); float* attn = AF(32L*256); float* li = AF(32L*1024);
  float* lgt  = AF((long)992*50000);

  auto PACK = [&](const float* s, int lds, int scol, int K, u16* d, int ldd, int dcol, int Kp, int R){
    long total = (long)R*Kp;
    pack_bf16_k<<<(int)((total+255)/256), 256, 0, stream>>>(s,lds,scol,K,d,ldd,dcol,Kp,total);
  };
  auto GEMM = [&](const u16* A, long sA, const u16* W, long sW, const float* bias,
                  float* C, long sC, int Mr, int N, int Kp, int act, int bat){
    dim3 g((N+511)/512, Mr/32, bat);
    if ((N & 63) == 0)
      gemm_bf16_nt<false><<<g, 256, 0, stream>>>(A,sA,W,sW,bias,C,sC,Mr,N,Kp,act);
    else
      gemm_bf16_nt<true ><<<g, 256, 0, stream>>>(A,sA,W,sW,bias,C,sC,Mr,N,Kp,act);
  };
  auto ADD = [&](const float* a, const float* b, float* d, int n){
    add_vec_k<<<(n+255)/256, 256, 0, stream>>>(a,b,d,n);
  };
  auto COPYC = [&](const float* s, int K, float* d, int ldd, int coloff, int R){
    long total = (long)R*K;
    copy_cols_k<<<(int)((total+255)/256), 256, 0, stream>>>(s,K,d,ldd,coloff,total);
  };

  // ---- bias sums
  ADD(F[10],F[11],bs0f,2048); ADD(F[14],F[15],bs0b,2048);
  ADD(F[18],F[19],bs1f,2048); ADD(F[22],F[23],bs1b,2048);
  ADD(F[34],F[35],bsd0,4096); ADD(F[38],F[39],bsd1,4096);
  // ---- weight packs (fp32 -> bf16, K padded to /32)
  PACK(F[8], 503,0,503, Wih0f,512,0,512, 2048);  PACK(F[9], 512,0,512, Whh0f,512,0,512, 2048);
  PACK(F[12],503,0,503, Wih0b,512,0,512, 2048);  PACK(F[13],512,0,512, Whh0b,512,0,512, 2048);
  PACK(F[16],1024,0,1024, Wih1f,1024,0,1024, 2048); PACK(F[17],512,0,512, Whh1f,512,0,512, 2048);
  PACK(F[20],1024,0,1024, Wih1b,1024,0,1024, 2048); PACK(F[21],512,0,512, Whh1b,512,0,512, 2048);
  PACK(F[24],1024,0,1024, ltWp,1024,0,1024, 1024);
  PACK(F[26],2048,0,2048, updWp,2048,0,2048, 1024);
  PACK(F[27],2048,0,2048, gateWp,2048,0,2048, 1024);
  PACK(F[28],1024,0,1024, enctrWp,1024,0,1024, 1024);
  PACK(F[30],1324,0,300,  redWp,1344,0,320, 300);
  PACK(F[30],1324,300,1024, redWp,1344,320,1024, 300);
  PACK(F[32],300,0,300,   Wc0p,1344,0,320, 4096);
  PACK(F[33],1024,0,1024, Wc0p,1344,320,1024, 4096);
  PACK(F[36],1024,0,1024, Wc1p,2048,0,1024, 4096);
  PACK(F[37],1024,0,1024, Wc1p,2048,1024,1024, 4096);
  PACK(F[40],2048,0,2048, catWp,2048,0,2048, 1024);
  PACK(F[42],1024,0,1024, logWp,1024,0,1024, 50000);

  // ---- graph attention + embedding assembly -> emb_bf [8192,512]
  graph_emb_k<<<8192, 128, 0, stream>>>(F[0],F[2],F[3],F[4],F[5],F[6],F[7],
                                        src,tag,cs,msk, emb_bf);

  // ---- encoder layer 0
  GEMM(emb_bf,0, Wih0f,0, bs0f, xz_a,0, 8192,2048,512,0,1);
  GEMM(emb_bf,0, Wih0b,0, bs0b, xz_b,0, 8192,2048,512,0,1);
  lstm_scan_k<<<1,256,0,stream>>>(xz_a, Whh0f, y0, 0,   zbuf, h0ff,c0ff, 256, 0);
  lstm_scan_k<<<1,256,0,stream>>>(xz_b, Whh0b, y0, 512, zbuf, h0bf,c0bf, 256, 1);
  PACK(y0,1024,0,1024, y0_bf,1024,0,1024, 8192);
  // ---- encoder layer 1
  GEMM(y0_bf,0, Wih1f,0, bs1f, xz_a,0, 8192,2048,1024,0,1);
  GEMM(y0_bf,0, Wih1b,0, bs1b, xz_b,0, 8192,2048,1024,0,1);
  lstm_scan_k<<<1,256,0,stream>>>(xz_a, Whh1f, y1, 0,   zbuf, h1ff,c1ff, 256, 0);
  lstm_scan_k<<<1,256,0,stream>>>(xz_b, Whh1b, y1, 512, zbuf, h1bf,c1bf, 256, 1);
  PACK(y1,1024,0,1024, out_bf,1024,0,1024, 8192);

  // ---- gated self attention
  GEMM(out_bf,0, ltWp,0, F[25], mem,0, 8192,1024,1024,0,1);
  PACK(mem,1024,0,1024, mem_bf,1024,0,1024, 8192);
  GEMM(out_bf,(long)256*1024, mem_bf,(long)256*1024, nullptr, en,(long)256*256,
       256,256,1024,0,32);
  softmax_k<<<8192,256,0,stream>>>(en, en, 256);
  PACK(en,256,0,256, sc_bf,256,0,256, 8192);
  transpose_pack_k<<<32768,256,0,stream>>>(y1, outT_bf);
  GEMM(sc_bf,(long)256*256, outT_bf,(long)1024*256, nullptr, ctx,(long)256*1024,
       256,1024,256,0,32);
  PACK(y1,1024,0,1024,  gin_bf,2048,0,1024, 8192);
  PACK(ctx,1024,0,1024, gin_bf,2048,1024,1024, 8192);
  GEMM(gin_bf,0, updWp,0,  nullptr, ft,0, 8192,1024,2048,0,1);
  GEMM(gin_bf,0, gateWp,0, nullptr, gt,0, 8192,1024,2048,0,1);
  combine_k<<<32768,256,0,stream>>>(ft, gt, y1, eo);
  PACK(eo,1024,0,1024, eo_bf,1024,0,1024, 8192);
  GEMM(eo_bf,0, enctrWp,0, F[29], mems,0, 8192,1024,1024,0,1);

  // ---- decoder init
  COPYC(h0ff,512, dh0,1024,0,32);  COPYC(h0bf,512, dh0,1024,512,32);
  COPYC(h1ff,512, dh1,1024,0,32);  COPYC(h1bf,512, dh1,1024,512,32);
  COPYC(c0ff,512, dc0,1024,0,32);  COPYC(c0bf,512, dc0,1024,512,32);
  COPYC(c1ff,512, dc1,1024,0,32);  COPYC(c1bf,512, dc1,1024,512,32);
  fill_k<<<128,256,0,stream>>>(pctx, 32L*1024, 0.0f);

  // ---- decoder scan (31 steps)
  for (int st=0; st<31; ++st){
    pack_dec_in_k<<<(32*1344+255)/256,256,0,stream>>>(F[1], trg, st, 32, pctx, xred_bf);
    GEMM(xred_bf,0, redWp,0, F[31], xr,0, 32,300,1344,0,1);
    PACK(xr,300,0,300,    xh0_bf,1344,0,320, 32);
    PACK(dh0,1024,0,1024, xh0_bf,1344,320,1024, 32);
    GEMM(xh0_bf,0, Wc0p,0, bsd0, zdec,0, 32,4096,1344,0,1);
    lstm_gate_k<<<128,256,0,stream>>>(zdec, dc0, dh0, 1024, 32*1024);
    PACK(dh0,1024,0,1024, xh1_bf,2048,0,1024, 32);
    PACK(dh1,1024,0,1024, xh1_bf,2048,1024,1024, 32);
    GEMM(xh1_bf,0, Wc1p,0, bsd1, zdec,0, 32,4096,2048,0,1);
    lstm_gate_k<<<128,256,0,stream>>>(zdec, dc1, dh1, 1024, 32*1024);
    energy_k<<<1024,256,0,stream>>>(dh1, mems, eAll + (long)st*32*256);
    softmax_k<<<32,256,0,stream>>>(eAll + (long)st*32*256, attn, 256);
    ctx2_k<<<128,256,0,stream>>>(attn, mems, pctx);
    PACK(dh1,1024,0,1024,  hc_bf,2048,0,1024, 32);
    PACK(pctx,1024,0,1024, hc_bf,2048,1024,1024, 32);
    GEMM(hc_bf,0, catWp,0, F[41], li,0, 32,1024,2048,1,1);
    PACK(li,1024,0,1024, Li_bf + (long)st*32*1024, 1024,0,1024, 32);
  }

  // ---- batched logits GEMM + pointer scatter-max assembly
  GEMM(Li_bf,0, logWp,0, F[43], lgt,0, 992,50000,1024,0,1);
  long total = 992L*50050;
  int blk = (int)((total+255)/256);
  extend_k<<<blk,256,0,stream>>>(lgt, out, total);
  scat_k<<<992,256,0,stream>>>(eAll, ext, out);
  final_k<<<blk,256,0,stream>>>(out, total);
}
